// SelfAtten_16862041604642
// MI455X (gfx1250) — compile-verified, matrix operand order checked by reading, not run on hardware
//
#include <hip/hip_runtime.h>
#include <hip/hip_bf16.h>
#include <math.h>

#define BB 4
#define SS 2048
#define EE 1024
#define DD 512

typedef __attribute__((ext_vector_type(16))) __bf16        bf16x16;
typedef __attribute__((ext_vector_type(8)))  float         f32x8;
typedef __attribute__((ext_vector_type(4)))  unsigned int  u32x4;

// f32 -> bf16 round-to-nearest-even
__device__ __forceinline__ unsigned short f2bf(float f) {
  unsigned int x = __builtin_bit_cast(unsigned int, f);
  x += 0x7FFFu + ((x >> 16) & 1u);
  return (unsigned short)(x >> 16);
}

union FragU { bf16x16 f; u32x4 h[2]; };

// WMMA 16-bit operand fragment = 16 bf16 per lane = two 16-byte loads
__device__ __forceinline__ bf16x16 ld_frag(const unsigned short* p0,
                                           const unsigned short* p1) {
  FragU u;
  u.h[0] = *(const u32x4*)p0;
  u.h[1] = *(const u32x4*)p1;
  return u.f;
}

// Async DMA: 16 bytes per lane, global -> LDS, tracked by ASYNCcnt.
// lds  : LDS byte offset (per-lane VGPR)
// voff : 32-bit byte offset added to the uniform 64-bit base (GVS mode)
__device__ __forceinline__ void async_cp16(unsigned int lds,
                                           unsigned int voff,
                                           const void* sbase) {
  asm volatile("global_load_async_to_lds_b128 %0, %1, %2"
               :: "v"(lds), "v"(voff), "s"(sbase)
               : "memory");
}

__device__ __forceinline__ unsigned int lds_addr(const void* p) {
  // generic LDS pointer: low 32 bits are the LDS byte offset
  return (unsigned int)(unsigned long long)(uintptr_t)p;
}

// ---------------------------------------------------------------------------
// One-time precision conversion: X f32 -> bf16 (same layout, packed stores)
// ---------------------------------------------------------------------------
__global__ __launch_bounds__(256) void cvt_x(const float* __restrict__ X,
                                             unsigned short* __restrict__ Xb) {
  size_t i = (size_t)blockIdx.x * 256 + threadIdx.x;  // one dword pair each
  float2 v = ((const float2*)X)[i];
  unsigned int pk = (unsigned int)f2bf(v.x) | ((unsigned int)f2bf(v.y) << 16);
  ((unsigned int*)Xb)[i] = pk;
}

// W[E][D] f32 -> Wt[D][E] bf16 (pre-transposed so GEMM B-tiles are row copies)
__global__ __launch_bounds__(256) void cvt_wT(const float* __restrict__ W,
                                              unsigned short* __restrict__ Wt) {
  int i = blockIdx.x * 256 + threadIdx.x;  // EE*DD elements
  int k = i >> 9;          // / DD
  int n = i & (DD - 1);
  Wt[(size_t)n * EE + k] = f2bf(W[i]);     // read coalesced, write strided (1MB, one-time)
}

// ---------------------------------------------------------------------------
// Kernel 1: Y = Xb[8192,1024] * Wt^T + bias, bf16 in, bf16 out.
// is_vt==0 : dst [b][s][d] (Q, K);  is_vt==1 : dst [b][d][s] (V transposed)
// 128x128 tile, K-step 32, double-buffered LDS filled by async DMA.
// ---------------------------------------------------------------------------
__global__ __launch_bounds__(256) void qkv_gemm(
    const unsigned short* __restrict__ Xb, const unsigned short* __restrict__ Wt,
    const float* __restrict__ bias, unsigned short* __restrict__ dst,
    int is_vt) {
  __shared__ unsigned short As[2][128 * 32];  // [m][k] bf16, 8KB per buffer
  __shared__ unsigned short Bs[2][128 * 32];  // [n][k] bf16
  __shared__ unsigned short Cs[128 * 130];    // padded epilogue staging

  const int tid  = threadIdx.x;
  const int lane = tid & 31;
  const int ln   = lane & 15;
  const int half = lane >> 4;
  const int w    = tid >> 5;
  const int wm   = (w >> 1) * 32;
  const int wn   = (w & 1) * 64;
  const int bs0  = blockIdx.x * 128;
  const int n0   = blockIdx.y * 128;

  // per-thread DMA chunk geometry: 128 rows x 64B tile = 512 x 16B chunks,
  // thread handles chunks {tid, tid+256} for As and for Bs.
  const int rowc = tid >> 2;            // 0..63
  const int part = tid & 3;
  const unsigned int voff0 = (unsigned)rowc * (EE * 2) + part * 16;
  const unsigned int voff1 = voff0 + 64 * (EE * 2);
  const unsigned int ldsA  = lds_addr(&As[0][0]) + rowc * 64 + part * 16;
  const unsigned int ldsB  = lds_addr(&Bs[0][0]) + rowc * 64 + part * 16;

  const char* xrow = (const char*)Xb + (size_t)bs0 * EE * 2;
  const char* wrow = (const char*)Wt + (size_t)n0 * EE * 2;

  f32x8 acc[2][4];
#pragma unroll
  for (int mt = 0; mt < 2; ++mt)
#pragma unroll
    for (int nt = 0; nt < 4; ++nt) acc[mt][nt] = {};

  // prologue: fill buffer 0 (k=0)
  {
    const void* xb = xrow;  // e0 = 0
    const void* wb = wrow;
    async_cp16(ldsA, voff0, xb);
    async_cp16(ldsA + 4096, voff1, xb);
    async_cp16(ldsB, voff0, wb);
    async_cp16(ldsB + 4096, voff1, wb);
  }

  for (int ks = 0; ks < 32; ++ks) {
    const int cur = ks & 1;
    if (ks < 31) {
      const int e1 = (ks + 1) * 32;
      const void* xb = xrow + (size_t)e1 * 2;
      const void* wb = wrow + (size_t)e1 * 2;
      const unsigned int bo = (cur ^ 1) * 8192;
      async_cp16(ldsA + bo, voff0, xb);
      async_cp16(ldsA + bo + 4096, voff1, xb);
      async_cp16(ldsB + bo, voff0, wb);
      async_cp16(ldsB + bo + 4096, voff1, wb);
      asm volatile("s_wait_asynccnt 0x4" ::: "memory");  // prev batch landed
    } else {
      asm volatile("s_wait_asynccnt 0x0" ::: "memory");
    }
    __syncthreads();

#pragma unroll
    for (int mt = 0; mt < 2; ++mt) {
      int m = wm + mt * 16 + ln;
      bf16x16 a = ld_frag(&As[cur][m * 32 + 8 * half],
                          &As[cur][m * 32 + 16 + 8 * half]);
#pragma unroll
      for (int nt = 0; nt < 4; ++nt) {
        int n = wn + nt * 16 + ln;
        bf16x16 b = ld_frag(&Bs[cur][n * 32 + 8 * half],
                            &Bs[cur][n * 32 + 16 + 8 * half]);
        acc[mt][nt] = __builtin_amdgcn_wmma_f32_16x16x32_bf16(
            false, a, false, b, (short)0, acc[mt][nt], false, false);
      }
    }
    __syncthreads();  // reads done before next iteration's DMA overwrites
  }

  // bias add + spill C to LDS (f32 C layout: VGPR r -> M=r+8*half, N=lane&15)
#pragma unroll
  for (int nt = 0; nt < 4; ++nt) {
    float bv = bias[n0 + wn + nt * 16 + ln];
#pragma unroll
    for (int mt = 0; mt < 2; ++mt) {
#pragma unroll
      for (int r = 0; r < 8; ++r) {
        int row = wm + mt * 16 + r + 8 * half;
        Cs[row * 130 + wn + nt * 16 + ln] = f2bf(acc[mt][nt][r] + bv);
      }
    }
  }
  __syncthreads();

  if (!is_vt) {
    for (int i = tid; i < 128 * 128; i += 256) {
      int r = i >> 7, c = i & 127;  // coalesced along d
      dst[(size_t)(bs0 + r) * DD + n0 + c] = Cs[r * 130 + c];
    }
  } else {
    int bat = bs0 >> 11;
    int s0  = bs0 & (SS - 1);
    for (int i = tid; i < 128 * 128; i += 256) {
      int r = i & 127, c = i >> 7;  // coalesced along s
      dst[(size_t)bat * DD * SS + (size_t)(n0 + c) * SS + s0 + r] =
          Cs[r * 130 + c];
    }
  }
}

// ---------------------------------------------------------------------------
// Kernel 2: causal flash attention, Q[b][s][d], K[b][s][d], Vt[b][d][s] bf16.
// WG = (16-query block, batch); 8 waves; cross-wave online softmax in LDS.
// ---------------------------------------------------------------------------
__global__ __launch_bounds__(256) void flash_attn(
    const unsigned short* __restrict__ Qb, const unsigned short* __restrict__ Kb,
    const unsigned short* __restrict__ Vt, float* __restrict__ Out) {
  __shared__ unsigned short Qs[16 * 512];
  __shared__ unsigned short Ps[16 * 128];
  __shared__ float redm[8 * 16];
  __shared__ float reds[8 * 16];
  __shared__ float mstate[16], lstate[16], fac[16];

  const int tid  = threadIdx.x;
  const int lane = tid & 31;
  const int ln   = lane & 15;
  const int half = lane >> 4;
  const int w    = tid >> 5;
  const int q0   = blockIdx.x * 16;
  const int bat  = blockIdx.y;
  const size_t qkbase = (size_t)bat * SS * DD;
  const size_t vbase  = (size_t)bat * DD * SS;

  // async DMA Q tile (16KB) into LDS: 4 x 16B chunks per thread
  {
    const void* src = (const void*)(Qb + qkbase + (size_t)q0 * DD);
    unsigned int qsb = lds_addr(&Qs[0]);
#pragma unroll
    for (int c = 0; c < 4; ++c) {
      unsigned int off = (unsigned)(tid + c * 256) * 16;
      async_cp16(qsb + off, off, src);
    }
    asm volatile("s_wait_asynccnt 0x0" ::: "memory");
  }
  if (tid < 16) { mstate[tid] = -INFINITY; lstate[tid] = 0.f; }
  __syncthreads();

  f32x8 accO[4];
#pragma unroll
  for (int nt = 0; nt < 4; ++nt) accO[nt] = {};

  const float scale = 0.044194173824159216f;  // 1/sqrt(512)
  const int q_hi = q0 + 15;
  const int nit  = (q_hi >> 7) + 1;           // causal: only kv0 <= q_hi

  for (int j = 0; j < nit; ++j) {
    const int kv0 = j * 128;
    const int kvr = kv0 + w * 16;

    // scores S_w = Q(16x512) * K_tile^T, f32 accum
    f32x8 s = {};
#pragma unroll
    for (int kc = 0; kc < 16; ++kc) {
      bf16x16 a = ld_frag(&Qs[ln * 512 + kc * 32 + 8 * half],
                          &Qs[ln * 512 + kc * 32 + 16 + 8 * half]);
      const unsigned short* kp =
          Kb + qkbase + (size_t)(kvr + ln) * DD + kc * 32 + 8 * half;
      bf16x16 b = ld_frag(kp, kp + 16);
      s = __builtin_amdgcn_wmma_f32_16x16x32_bf16(false, a, false, b, (short)0,
                                                  s, false, false);
    }

    // scale, causal mask, wave-level row max over 16-lane halves
    float rm[8];
#pragma unroll
    for (int r = 0; r < 8; ++r) {
      int qrow = q0 + r + 8 * half;
      int kcol = kvr + ln;
      float v = s[r] * scale;
      if (kcol > qrow) v = -INFINITY;
      s[r] = v;
      float t = v;
      t = fmaxf(t, __shfl_xor(t, 1, 32));
      t = fmaxf(t, __shfl_xor(t, 2, 32));
      t = fmaxf(t, __shfl_xor(t, 4, 32));
      t = fmaxf(t, __shfl_xor(t, 8, 32));
      rm[r] = t;
    }
    if (ln == 0) {
#pragma unroll
      for (int r = 0; r < 8; ++r) redm[w * 16 + r + 8 * half] = rm[r];
    }
    __syncthreads();

    // cross-wave running max + rescale factor
    if (tid < 16) {
      float mo = mstate[tid];
      float mj = redm[tid];
#pragma unroll
      for (int wv = 1; wv < 8; ++wv) mj = fmaxf(mj, redm[wv * 16 + tid]);
      float mn = fmaxf(mo, mj);
      mstate[tid] = mn;
      fac[tid] = expf(mo - mn);  // first iter: exp(-inf - mn) = 0
    }
    __syncthreads();

    // p = exp(s - m), row sums, stage P (bf16) in LDS, rescale O
    float mrow[8], frow[8];
#pragma unroll
    for (int r = 0; r < 8; ++r) {
      mrow[r] = mstate[r + 8 * half];
      frow[r] = fac[r + 8 * half];
    }
    float rs[8];
#pragma unroll
    for (int r = 0; r < 8; ++r) {
      float p = expf(s[r] - mrow[r]);
      Ps[(r + 8 * half) * 128 + w * 16 + ln] = f2bf(p);
      float t = p;
      t += __shfl_xor(t, 1, 32);
      t += __shfl_xor(t, 2, 32);
      t += __shfl_xor(t, 4, 32);
      t += __shfl_xor(t, 8, 32);
      rs[r] = t;
    }
    if (ln == 0) {
#pragma unroll
      for (int r = 0; r < 8; ++r) reds[w * 16 + r + 8 * half] = rs[r];
    }
#pragma unroll
    for (int nt = 0; nt < 4; ++nt)
#pragma unroll
      for (int r = 0; r < 8; ++r) accO[nt][r] *= frow[r];
    __syncthreads();

    // l update (16 threads) alongside P*V (all waves)
    if (tid < 16) {
      float l = lstate[tid] * fac[tid];
#pragma unroll
      for (int wv = 0; wv < 8; ++wv) l += reds[wv * 16 + tid];
      lstate[tid] = l;
    }
#pragma unroll
    for (int kc = 0; kc < 4; ++kc) {
      bf16x16 a = ld_frag(&Ps[ln * 128 + kc * 32 + 8 * half],
                          &Ps[ln * 128 + kc * 32 + 16 + 8 * half]);
#pragma unroll
      for (int nt = 0; nt < 4; ++nt) {
        int d = w * 64 + nt * 16 + ln;
        const unsigned short* vp =
            Vt + vbase + (size_t)d * SS + kv0 + kc * 32 + 8 * half;
        bf16x16 b = ld_frag(vp, vp + 16);
        accO[nt] = __builtin_amdgcn_wmma_f32_16x16x32_bf16(
            false, a, false, b, (short)0, accO[nt], false, false);
      }
    }
    __syncthreads();
  }

  // normalize and store O (f32, coalesced across lanes)
  float inv[8];
#pragma unroll
  for (int r = 0; r < 8; ++r) inv[r] = 1.0f / lstate[r + 8 * half];
#pragma unroll
  for (int nt = 0; nt < 4; ++nt) {
    int d = w * 64 + nt * 16 + ln;
#pragma unroll
    for (int r = 0; r < 8; ++r) {
      Out[((size_t)bat * SS + q0 + r + 8 * half) * DD + d] = accO[nt][r] * inv[r];
    }
  }
}

// ---------------------------------------------------------------------------
extern "C" void kernel_launch(void* const* d_in, const int* in_sizes, int n_in,
                              void* d_out, int out_size, void* d_ws,
                              size_t ws_size, hipStream_t stream) {
  (void)in_sizes; (void)n_in; (void)out_size; (void)ws_size;
  const float* x  = (const float*)d_in[0];
  const float* wq = (const float*)d_in[1];
  const float* bq = (const float*)d_in[2];
  const float* wk = (const float*)d_in[3];
  const float* bk = (const float*)d_in[4];
  const float* wv = (const float*)d_in[5];
  const float* bv = (const float*)d_in[6];
  float* out = (float*)d_out;

  // workspace layout (bf16): Q/K [b][s][d], V^T [b][d][s], Xb [bs][e], Wt [d][e] x3
  unsigned short* Qb  = (unsigned short*)d_ws;
  unsigned short* Kb  = Qb + (size_t)BB * SS * DD;
  unsigned short* Vt  = Kb + (size_t)BB * SS * DD;
  unsigned short* Xb  = Vt + (size_t)BB * SS * DD;
  unsigned short* WtQ = Xb + (size_t)BB * SS * EE;
  unsigned short* WtK = WtQ + (size_t)EE * DD;
  unsigned short* WtV = WtK + (size_t)EE * DD;

  dim3 blk(256);
  cvt_x<<<dim3((BB * SS * EE) / 512), blk, 0, stream>>>(x, Xb);
  cvt_wT<<<dim3((EE * DD) / 256), blk, 0, stream>>>(wq, WtQ);
  cvt_wT<<<dim3((EE * DD) / 256), blk, 0, stream>>>(wk, WtK);
  cvt_wT<<<dim3((EE * DD) / 256), blk, 0, stream>>>(wv, WtV);

  dim3 g1(BB * SS / 128, DD / 128);  // 64 x 4
  qkv_gemm<<<g1, blk, 0, stream>>>(Xb, WtQ, bq, Qb, 0);
  qkv_gemm<<<g1, blk, 0, stream>>>(Xb, WtK, bk, Kb, 0);
  qkv_gemm<<<g1, blk, 0, stream>>>(Xb, WtV, bv, Vt, 1);

  dim3 g2(SS / 16, BB);  // 128 x 4
  flash_attn<<<g2, blk, 0, stream>>>(Qb, Kb, Vt, out);
}
